// ODE2VAE_69965017252422
// MI455X (gfx1250) — matrix operand order repaired and checked by reading, not compiled, not run on hardware
//
#include <hip/hip_runtime.h>
#include <hip/hip_bf16.h>
#include <math.h>

// ---------------- CDNA5 WMMA types ----------------
typedef __attribute__((ext_vector_type(16))) __bf16          v16bf;
typedef __attribute__((ext_vector_type(8)))  float           v8f;
typedef __attribute__((ext_vector_type(8)))  unsigned short  v8us;

#define NBATCH 512
#define TSTEPS 16
#define LMC    2
#define QDIM   8
#define HD     512
#define MTOT   (LMC*NBATCH*TSTEPS)      // 16384 decoded samples
#define DTSTEP 0.1f
#define LOG2PI_F 1.83787706640934548356f

// =====================================================================
// Encoder: 3 direct convs per sample (tiny FLOPs), block = 1 sample.
// =====================================================================
__global__ __launch_bounds__(256)
void k_encode(const float* __restrict__ X,
              const float* __restrict__ c1w, const float* __restrict__ c1b,
              const float* __restrict__ c2w, const float* __restrict__ c2b,
              const float* __restrict__ c3w, const float* __restrict__ c3b,
              float* __restrict__ h_enc) {
  __shared__ float xb[28*28];
  __shared__ float b1[8*14*14];
  __shared__ float b2[16*7*7];
  const int n = blockIdx.x, tid = threadIdx.x;
  const float* xp = X + (size_t)n * TSTEPS * 784;   // frame t=0
  for (int i = tid; i < 784; i += 256) xb[i] = xp[i];
  __syncthreads();
  for (int idx = tid; idx < 8*196; idx += 256) {
    int co = idx / 196, p = idx - co*196, oy = p / 14, ox = p - oy*14;
    float acc = c1b[co];
    for (int ky = 0; ky < 5; ++ky) {
      int iy = oy*2 + ky - 2; if ((unsigned)iy >= 28u) continue;
      for (int kx = 0; kx < 5; ++kx) {
        int ix = ox*2 + kx - 2; if ((unsigned)ix >= 28u) continue;
        acc += xb[iy*28+ix] * c1w[co*25 + ky*5 + kx];
      }
    }
    b1[idx] = fmaxf(acc, 0.f);
  }
  __syncthreads();
  for (int idx = tid; idx < 16*49; idx += 256) {
    int co = idx / 49, p = idx - co*49, oy = p / 7, ox = p - oy*7;
    float acc = c2b[co];
    for (int ci = 0; ci < 8; ++ci)
      for (int ky = 0; ky < 5; ++ky) {
        int iy = oy*2 + ky - 2; if ((unsigned)iy >= 14u) continue;
        for (int kx = 0; kx < 5; ++kx) {
          int ix = ox*2 + kx - 2; if ((unsigned)ix >= 14u) continue;
          acc += b1[ci*196 + iy*14 + ix] * c2w[(co*8+ci)*25 + ky*5 + kx];
        }
      }
    b2[idx] = fmaxf(acc, 0.f);
  }
  __syncthreads();
  for (int idx = tid; idx < 32*16; idx += 256) {
    int co = idx / 16, p = idx - co*16, oy = p / 4, ox = p - oy*4;
    float acc = c3b[co];
    for (int ci = 0; ci < 16; ++ci)
      for (int ky = 0; ky < 5; ++ky) {
        int iy = oy*2 + ky - 2; if ((unsigned)iy >= 7u) continue;
        for (int kx = 0; kx < 5; ++kx) {
          int ix = ox*2 + kx - 2; if ((unsigned)ix >= 7u) continue;
          acc += b2[ci*49 + iy*7 + ix] * c3w[(co*16+ci)*25 + ky*5 + kx];
        }
      }
    h_enc[(size_t)n*HD + idx] = fmaxf(acc, 0.f);
  }
}

// =====================================================================
// Latent heads: qz0_m, qz0_logv, z0 = m + eps*exp(logv)
// =====================================================================
__global__ void k_latent(const float* __restrict__ h_enc,
                         const float* __restrict__ fc1w, const float* __restrict__ fc1b,
                         const float* __restrict__ fc2w, const float* __restrict__ fc2b,
                         const float* __restrict__ eps,
                         float* __restrict__ outQm, float* __restrict__ outQlv,
                         float* __restrict__ z0w) {
  int t = blockIdx.x*blockDim.x + threadIdx.x;
  if (t >= NBATCH*2*QDIM) return;
  int n = t >> 4, j = t & 15;
  const float* h = h_enc + (size_t)n*HD;
  float m = fc1b[j], lv = fc2b[j];
  for (int k = 0; k < HD; ++k) { float hv = h[k]; m += hv*fc1w[j*HD+k]; lv += hv*fc2w[j*HD+k]; }
  outQm[t] = m; outQlv[t] = lv;
  z0w[t]   = m + eps[t]*expf(lv);
}

__global__ void k_logp0(const float* __restrict__ eps, float* __restrict__ logp0w) {
  int n = blockIdx.x*blockDim.x + threadIdx.x;
  if (n >= NBATCH) return;
  float s = 0.f;
  for (int d = 0; d < 16; ++d) { float e = eps[n*16+d]; s += e*e; }
  logp0w[n] = -0.5f*s - 8.f*LOG2PI_F;
}

// =====================================================================
// RK4 integration of augmented state (z[16], logp) with exact divergence.
// One 64-thread block per (l, n). Writes ztL straight to d_out.
// =====================================================================
__global__ __launch_bounds__(64)
void k_ode(const float* __restrict__ z0w, const float* __restrict__ logp0w,
           const float* __restrict__ mW0, const float* __restrict__ mb0,
           const float* __restrict__ mW1, const float* __restrict__ mb1,
           const float* __restrict__ mW2, const float* __restrict__ mb2,
           float* __restrict__ ztOut, float* __restrict__ logpLw) {
  __shared__ float zfull[17];
  __shared__ float zs[16];
  __shared__ float h1[50], h2v[50], d2v[50];
  __shared__ float Aq[50*8], Gq[50*8];
  __shared__ float kst[4][17];
  __shared__ float red[8];
  const int tid = threadIdx.x;
  const int l = blockIdx.x / NBATCH, n = blockIdx.x - l*NBATCH;
  const size_t mbase = (size_t)(l*NBATCH + n) * TSTEPS;

  if (tid < 16) zfull[tid] = z0w[n*16 + tid];
  if (tid == 16) zfull[16] = logp0w[n];
  __syncthreads();
  if (tid < 16) ztOut[mbase*16 + tid] = zfull[tid];
  if (tid == 16) logpLw[mbase] = zfull[16];

  const float cc[4] = {0.f, DTSTEP*0.5f, DTSTEP*0.5f, DTSTEP};
  for (int step = 0; step < TSTEPS-1; ++step) {
    for (int st = 0; st < 4; ++st) {
      if (tid < 16) zs[tid] = (st == 0) ? zfull[tid] : zfull[tid] + cc[st]*kst[st-1][tid];
      __syncthreads();
      if (tid < 50) {                                   // layer 0 + A = D1*W0[:, :Q]
        float a = mb0[l*50 + tid];
        const float* w0 = mW0 + (size_t)(l*50 + tid)*16;
        for (int j = 0; j < 16; ++j) a += w0[j]*zs[j];
        float h = tanhf(a); h1[tid] = h;
        float d = 1.f - h*h;
        for (int q = 0; q < 8; ++q) Aq[tid*8+q] = d * w0[q];
      }
      __syncthreads();
      if (tid < 50) {                                   // layer 1 + G = W1*A
        float a = mb1[l*50 + tid];
        float g[8] = {0,0,0,0,0,0,0,0};
        const float* w1 = mW1 + (size_t)(l*50 + tid)*50;
        for (int j = 0; j < 50; ++j) {
          float w = w1[j]; a += w*h1[j];
          for (int q = 0; q < 8; ++q) g[q] += w*Aq[j*8+q];
        }
        float h = tanhf(a); h2v[tid] = h; d2v[tid] = 1.f - h*h;
        for (int q = 0; q < 8; ++q) Gq[tid*8+q] = g[q];
      }
      __syncthreads();
      if (tid < 8) {                                    // layer 2 + trace partial
        float f = mb2[l*8 + tid], tq = 0.f;
        const float* w2 = mW2 + (size_t)(l*8 + tid)*50;
        for (int j = 0; j < 50; ++j) { float w = w2[j]; f += w*h2v[j]; tq += w*d2v[j]*Gq[j*8+tid]; }
        kst[st][tid] = f; red[tid] = tq;
      } else if (tid < 16) {
        kst[st][tid] = zs[tid-8];                       // ds/dt = v
      }
      __syncthreads();
      if (tid == 0) { float tr = 0.f; for (int q = 0; q < 8; ++q) tr += red[q]; kst[st][16] = -tr; }
      __syncthreads();
    }
    if (tid < 17)
      zfull[tid] += (DTSTEP/6.f)*(kst[0][tid] + 2.f*kst[1][tid] + 2.f*kst[2][tid] + kst[3][tid]);
    __syncthreads();
    size_t mi = mbase + step + 1;
    if (tid < 16) ztOut[mi*16 + tid] = zfull[tid];
    if (tid == 16) logpLw[mi] = zfull[16];
    __syncthreads();
  }
}

// =====================================================================
// Decoder fc3: h[m, j] = s[m, :] . fc3_w[j, :] + b  -> bf16 act (as [32,4,4])
// =====================================================================
__global__ void k_fcdec(const float* __restrict__ zt, const float* __restrict__ fc3w,
                        const float* __restrict__ fc3b, __bf16* __restrict__ acth) {
  int t = blockIdx.x*blockDim.x + threadIdx.x;
  if (t >= MTOT*HD) return;
  int m = t >> 9, j = t & 511;
  const float* s = zt + (size_t)m*16 + QDIM;
  float acc = fc3b[j];
  #pragma unroll
  for (int q = 0; q < 8; ++q) acc += s[q]*fc3w[j*8+q];
  acth[t] = (__bf16)acc;
}

// =====================================================================
// ConvTranspose weight -> TRANSPOSED GEMM B-matrix [Npad x Kpad] bf16.
// BmT[co][kk] = w[ci][co][k-1-ky][k-1-kx], kk = (ci*k+ky)*k+kx.
// =====================================================================
__global__ void k_wconv(const float* __restrict__ w, __bf16* __restrict__ BmT,
                        int Cout, int Ksz, int Ktot, int Kpad, int Npad, int CoutReal) {
  int t = blockIdx.x*blockDim.x + threadIdx.x;
  if (t >= Kpad*Npad) return;
  int co = t / Kpad, kk = t - co*Kpad;
  float v = 0.f;
  if (kk < Ktot && co < CoutReal) {
    int K2 = Ksz*Ksz;
    int ci = kk / K2, r = kk - ci*K2, ky = r / Ksz, kx = r - ky*Ksz;
    v = w[((size_t)(ci*Cout + co))*K2 + (Ksz-1-ky)*Ksz + (Ksz-1-kx)];
  }
  BmT[t] = (__bf16)v;
}

// =====================================================================
// Shared host/device geometry helpers
// =====================================================================
constexpr int cdiv_(int a, int b) { return (a + b - 1) / b; }
constexpr size_t convt_lds_bytes(int Cin, int IH, int Ksz, int STR, int PADL) {
  int K2 = Ksz*Ksz, Ktot = Cin*K2, Kpad = (Ktot + 31) & ~31;
  int DH = (IH - 1)*STR + 1 + 2*PADL;
  int CinP = cdiv_(Kpad, K2);
  int dilPad = ((CinP*DH*DH) + 7) & ~7;
  return (size_t)dilPad*2 + (size_t)Kpad*2;
}

// =====================================================================
// Implicit-GEMM ConvTranspose via bf16 WMMA, geometry as template consts.
// One block = one sample. LDS holds (a) a zero-padded, stride-dilated
// input image (all padding/parity/bounds pre-resolved -> branchless taps),
// and (b) a u16 K->offset table so the inner gather is:
//   two ds_load_b128 of offsets + 16x (add + ds_load_u16).
// B fragments are 32B-aligned v16bf vectors from L2-resident global.
// =====================================================================
template<int Cin, int IH, int Cout, int OH, int Ksz, int STR, int PADL, int FINAL>
__global__ __launch_bounds__(256)
void k_convt(const __bf16* __restrict__ actIn, const __bf16* __restrict__ BmT,
             const float* __restrict__ bias, __bf16* __restrict__ actOut,
             float* __restrict__ xrecOut, const float* __restrict__ Xin,
             float* __restrict__ accLhood) {
  constexpr int IW = IH, OW = OH, OHW = OH*OW;
  constexpr int K2 = Ksz*Ksz, Ktot = Cin*K2, Kpad = (Ktot + 31) & ~31;
  constexpr int Npad = (Cout + 15) & ~15, NT = Npad/16, MT = (OHW + 15)/16;
  constexpr int inElems = Cin*IH*IW;
  constexpr int DH = (IH - 1)*STR + 1 + 2*PADL, DW = DH, DHW = DH*DW;
  constexpr int CinP = cdiv_(Kpad, K2);             // extra zero channel absorbs K padding
  constexpr int dilElems = CinP*DHW;
  constexpr int dilPad = (dilElems + 7) & ~7;

  extern __shared__ __align__(32) char dynsm[];
  __bf16* dil = (__bf16*)dynsm;
  unsigned short* kOff = (unsigned short*)(dil + dilPad);

  const int sample = blockIdx.x;
  const int tid = threadIdx.x, lane = tid & 31, wid = tid >> 5;
  __builtin_prefetch(BmT, 0, 1);                    // global_prefetch_b8

  // Phase 1: zero dilated image + build K->LDS-offset table
  for (int i = tid; i < dilElems; i += 256) dil[i] = (__bf16)0.0f;
  for (int K = tid; K < Kpad; K += 256) {
    int ci = K / K2, r = K - ci*K2, ky = r / Ksz, kx = r - ky*Ksz;
    kOff[K] = (unsigned short)(ci*DHW + ky*DW + kx);
  }
  __syncthreads();
  // Phase 2: scatter input into dilated, padded positions
  for (int i = tid; i < inElems; i += 256) {
    int ci = i / (IH*IW), r = i - ci*IH*IW, iy = r / IW, ix = r - iy*IW;
    dil[(ci*DH + iy*STR + PADL)*DW + (ix*STR + PADL)] = actIn[(size_t)sample*inElems + i];
  }
  __syncthreads();

  const int Nlane = lane & 15;
  const int kHalfA = (lane & 16);                   // contiguous 16-bit A layout half-select
  const int kHalfB = (lane & 16);                   // contiguous 16-bit B layout half-select (same k order as A)

  for (int mt = wid; mt < MT; mt += 8) {
    v8f acc[NT];
    #pragma unroll
    for (int nt = 0; nt < NT; ++nt) acc[nt] = v8f{0.f,0.f,0.f,0.f,0.f,0.f,0.f,0.f};

    const int Mrow = mt*16 + Nlane;
    const int MrowC = (Mrow < OHW) ? Mrow : (OHW - 1);   // clamp, keep wave convergent
    const int oy = MrowC / OW, ox = MrowC - oy*OW;       // constant divisor
    const int pixBase = oy*DW + ox;                      // oy - PADL + ky + PADL

    for (int kc = 0; kc < Kpad; kc += 32) {
      v8us o0 = *(const v8us*)(kOff + kc + kHalfA);      // K = kc+kHalfA+0..7
      v8us o1 = *(const v8us*)(kOff + kc + kHalfA + 8);  // K = kc+kHalfA+8..15
      v16bf a;
      #pragma unroll
      for (int e = 0; e < 8; ++e) a[e]     = dil[pixBase + (int)o0[e]];
      #pragma unroll
      for (int e = 0; e < 8; ++e) a[8 + e] = dil[pixBase + (int)o1[e]];

      v16bf bf[NT];
      #pragma unroll
      for (int nt = 0; nt < NT; ++nt)
        bf[nt] = *(const v16bf*)(BmT + (size_t)(nt*16 + Nlane)*Kpad + kc + kHalfB);
      #pragma unroll
      for (int nt = 0; nt < NT; ++nt)
        acc[nt] = __builtin_amdgcn_wmma_f32_16x16x32_bf16(false, a, false, bf[nt],
                                                          (short)0, acc[nt], false, false);
    }

    // Epilogue: C/D layout — lane%16 = N, vgpr r -> M = r + 8*(lane>=16)
    #pragma unroll
    for (int nt = 0; nt < NT; ++nt) {
      const int Ncol = nt*16 + Nlane;
      const float bv = (Ncol < Cout) ? bias[Ncol] : 0.f;
      float lsum = 0.f;
      #pragma unroll
      for (int r = 0; r < 8; ++r) {
        int Mr = mt*16 + r + ((lane & 16) >> 1);
        if (Mr < OHW && Ncol < Cout) {
          float v = acc[nt][r] + bv;
          if (!FINAL) {
            actOut[(size_t)sample*Cout*OHW + (size_t)Ncol*OHW + Mr] = (__bf16)fmaxf(v, 0.f);
          } else {
            float sg = 1.f / (1.f + expf(-v));
            xrecOut[(size_t)sample*OHW + Mr] = sg;
            float xr = fminf(fmaxf(sg, 1e-6f), 1.f - 1e-6f);
            int nn = (sample >> 4) & (NBATCH-1);
            int tt = sample & (TSTEPS-1);
            float xv = Xin[((size_t)nn*TSTEPS + tt)*OHW + Mr];
            lsum += logf(xr)*xv + log1pf(-xr)*(1.f - xv);
          }
        }
      }
      if (FINAL) {                                   // wave reduction, then 1 atomic
        #pragma unroll
        for (int d = 16; d > 0; d >>= 1) lsum += __shfl_xor(lsum, d, 32);
        if (lane == 0) atomicAdd(accLhood, lsum);
      }
    }
  }
}

// =====================================================================
// KL reduction + final scalars
// =====================================================================
__global__ void k_kl(const float* __restrict__ zt, const float* __restrict__ logpLw,
                     float* __restrict__ accKl) {
  int m = blockIdx.x*blockDim.x + threadIdx.x;
  if (m >= MTOT) return;
  float s = 0.f;
  for (int d = 0; d < 16; ++d) { float z = zt[(size_t)m*16+d]; s += z*z; }
  float log_pzt = -0.5f*s - 8.f*LOG2PI_F;
  float val = logpLw[m] - log_pzt;
  #pragma unroll
  for (int d = 16; d > 0; d >>= 1) val += __shfl_xor(val, d, 32);
  if ((threadIdx.x & 31) == 0) atomicAdd(accKl, val);
}

__global__ void k_zero(float* __restrict__ acc) {
  if (threadIdx.x < 8) acc[threadIdx.x] = 0.f;
}

__global__ void k_final(const float* __restrict__ acc, float* __restrict__ outScal) {
  float lhood_m = acc[0] / (float)(LMC*NBATCH);
  float kl_m    = acc[1] / (float)(LMC*NBATCH);
  outScal[0] = lhood_m - kl_m;   // elbo
  outScal[1] = lhood_m;
  outScal[2] = kl_m;
}

// =====================================================================
// Host launcher
// =====================================================================
extern "C" void kernel_launch(void* const* d_in, const int* in_sizes, int n_in,
                              void* d_out, int out_size, void* d_ws, size_t ws_size,
                              hipStream_t stream) {
  const float* X    = (const float*)d_in[0];
  const float* c1w  = (const float*)d_in[1];  const float* c1b = (const float*)d_in[2];
  const float* c2w  = (const float*)d_in[3];  const float* c2b = (const float*)d_in[4];
  const float* c3w  = (const float*)d_in[5];  const float* c3b = (const float*)d_in[6];
  const float* fc1w = (const float*)d_in[7];  const float* fc1b= (const float*)d_in[8];
  const float* fc2w = (const float*)d_in[9];  const float* fc2b= (const float*)d_in[10];
  const float* fc3w = (const float*)d_in[11]; const float* fc3b= (const float*)d_in[12];
  const float* d1w  = (const float*)d_in[13]; const float* d1b = (const float*)d_in[14];
  const float* d2w  = (const float*)d_in[15]; const float* d2b = (const float*)d_in[16];
  const float* d3w  = (const float*)d_in[17]; const float* d3b = (const float*)d_in[18];
  const float* d4w  = (const float*)d_in[19]; const float* d4b = (const float*)d_in[20];
  const float* mW0  = (const float*)d_in[21]; const float* mb0 = (const float*)d_in[22];
  const float* mW1  = (const float*)d_in[23]; const float* mb1 = (const float*)d_in[24];
  const float* mW2  = (const float*)d_in[25]; const float* mb2 = (const float*)d_in[26];
  const float* eps  = (const float*)d_in[27];
  (void)in_sizes; (void)n_in; (void)out_size; (void)ws_size;  // L hardcoded = 2

  // ---- d_out layout: Xrec | qz0_m | qz0_logv | ztL | elbo,lhood_m,kl_m ----
  float* out      = (float*)d_out;
  float* outXrec  = out;
  float* outQm    = outXrec + (size_t)MTOT*784;         // 12,845,056
  float* outQlv   = outQm   + NBATCH*16;
  float* outZt    = outQlv  + NBATCH*16;
  float* outScal  = outZt   + (size_t)MTOT*16;

  // ---- workspace layout ----
  char* W = (char*)d_ws;
  size_t off = 0;
  auto ALLOC = [&](size_t bytes) -> char* {
    char* p = W + off; off = (off + bytes + 255) & ~(size_t)255; return p;
  };
  float*  h_enc  = (float*) ALLOC((size_t)NBATCH*HD*4);
  float*  z0w    = (float*) ALLOC((size_t)NBATCH*16*4);
  float*  logp0w = (float*) ALLOC((size_t)NBATCH*4);
  float*  logpLw = (float*) ALLOC((size_t)MTOT*4);
  float*  accw   = (float*) ALLOC(8*4);
  __bf16* B1     = (__bf16*)ALLOC((size_t)64*288 *2);   // [Npad][Kpad] transposed
  __bf16* B2     = (__bf16*)ALLOC((size_t)32*1600*2);
  __bf16* B3     = (__bf16*)ALLOC((size_t)16*800 *2);
  __bf16* B4     = (__bf16*)ALLOC((size_t)16*416 *2);
  __bf16* acth   = (__bf16*)ALLOC((size_t)MTOT*512*2);     // 32x4x4
  __bf16* a1     = (__bf16*)ALLOC((size_t)MTOT*64*36*2);   // 64x6x6
  __bf16* a2     = (__bf16*)ALLOC((size_t)MTOT*32*169*2);  // 32x13x13
  __bf16* a3     = (__bf16*)ALLOC((size_t)MTOT*16*784*2);  // 16x28x28

  k_zero<<<1, 32, 0, stream>>>(accw);

  // ---- encoder + latent + ODE ----
  k_encode<<<NBATCH, 256, 0, stream>>>(X, c1w, c1b, c2w, c2b, c3w, c3b, h_enc);
  k_latent<<<(NBATCH*16 + 255)/256, 256, 0, stream>>>(h_enc, fc1w, fc1b, fc2w, fc2b,
                                                      eps, outQm, outQlv, z0w);
  k_logp0<<<(NBATCH + 255)/256, 256, 0, stream>>>(eps, logp0w);
  k_ode<<<LMC*NBATCH, 64, 0, stream>>>(z0w, logp0w, mW0, mb0, mW1, mb1, mW2, mb2,
                                       outZt, logpLw);

  // ---- weight -> transposed bf16 B-matrix conversion ----
  k_wconv<<<(64*288  + 255)/256, 256, 0, stream>>>(d1w, B1, 64, 3, 288,  288,  64, 64);
  k_wconv<<<(32*1600 + 255)/256, 256, 0, stream>>>(d2w, B2, 32, 5, 1600, 1600, 32, 32);
  k_wconv<<<(16*800  + 255)/256, 256, 0, stream>>>(d3w, B3, 16, 5, 800,  800,  16, 16);
  k_wconv<<<(16*416  + 255)/256, 256, 0, stream>>>(d4w, B4, 1,  5, 400,  416,  16, 1);

  // ---- decoder ----
  k_fcdec<<<(MTOT*HD + 255)/256, 256, 0, stream>>>(outZt, fc3w, fc3b, acth);

  // d1: 32x4x4 -> 64x6x6, k3 s1 padl=2
  k_convt<32,4,64,6,3,1,2,0><<<MTOT, 256, convt_lds_bytes(32,4,3,1,2), stream>>>(
      acth, B1, d1b, a1, nullptr, nullptr, accw);
  // d2: 64x6x6 -> 32x13x13, k5 s2 padl=3
  k_convt<64,6,32,13,5,2,3,0><<<MTOT, 256, convt_lds_bytes(64,6,5,2,3), stream>>>(
      a1, B2, d2b, a2, nullptr, nullptr, accw);
  // d3: 32x13x13 -> 16x28x28, k5 s2 padl=3 (opad only affects OH)
  k_convt<32,13,16,28,5,2,3,0><<<MTOT, 256, convt_lds_bytes(32,13,5,2,3), stream>>>(
      a2, B3, d3b, a3, nullptr, nullptr, accw);
  // d4 (final): 16x28x28 -> 1x28x28, k5 s1 padl=2, sigmoid + BCE accumulate
  k_convt<16,28,1,28,5,1,2,1><<<MTOT, 256, convt_lds_bytes(16,28,5,1,2), stream>>>(
      a3, B4, d4b, nullptr, outXrec, X, accw);

  // ---- ELBO ----
  k_kl<<<(MTOT + 255)/256, 256, 0, stream>>>(outZt, logpLw, accw + 1);
  k_final<<<1, 1, 0, stream>>>(accw, outScal);
}
